// Decoder_72516227826046
// MI455X (gfx1250) — compile-verified, matrix operand order checked by reading, not run on hardware
//
#include <hip/hip_runtime.h>

#define B_TOT   512
#define NCELL   12
#define NNODE   40
#define EDIM    128
#define HDIM    128
#define ROWS    16
#define CLIPV   10.0f
#define NEGV    (-1e9f)

// output offsets (floats) in d_out, reference return order
#define OFF_CLP  0        // cell_log_prob [512]
#define OFF_NLP  512      // node_log_prob [512*12]
#define OFF_CRW  6656     // cell_reward   [512]
#define OFF_NRW  7168     // node_reward   [512*12]
#define OFF_CACT 13312    // cell_action   [512*12]
#define OFF_NACT 19456    // node_action   [512*12*40]

typedef __attribute__((ext_vector_type(16))) _Float16 v16h;
typedef __attribute__((ext_vector_type(8)))  _Float16 v8h;
typedef __attribute__((ext_vector_type(8)))  float    v8f;

// ---------------- WMMA fragment helpers (16x16x32 f16 -> f32) ----------------
// A fragment: 16xK row-major f16 in LDS.  ISA layout: lane l (0..15) holds row
// M=l, K=[0..7]+[16..23]; lane l+16 holds row l, K=[8..15]+[24..31].
__device__ __forceinline__ v16h lda_f16(const _Float16* A, int ld, int kbase) {
  int lane = threadIdx.x & 31;
  int row  = lane & 15;
  int ko   = kbase + ((lane >> 4) << 3);
  const _Float16* p = A + row * ld + ko;
  v8h lo = *(const v8h*)p;
  v8h hi = *(const v8h*)(p + 16);
  return __builtin_shufflevector(lo, hi, 0,1,2,3,4,5,6,7,8,9,10,11,12,13,14,15);
}

__device__ __forceinline__ v16h lda_f32(const float* A, int ld, int kbase) {
  int lane = threadIdx.x & 31;
  int row  = lane & 15;
  int ko   = kbase + ((lane >> 4) << 3);
  const float* p = A + row * ld + ko;
  v16h a;
#pragma unroll
  for (int i = 0; i < 8; ++i) { a[i] = (_Float16)p[i]; a[i + 8] = (_Float16)p[i + 16]; }
  return a;
}

// B fragment: W is [K x N] row-major fp32 in global; lane n holds column n.
__device__ __forceinline__ v16h ldb_g(const float* W, int ldn, int nbase, int kbase) {
  int lane = threadIdx.x & 31;
  int col  = nbase + (lane & 15);
  int ko   = kbase + ((lane >> 4) << 3);
  const float* p = W + (size_t)ko * ldn + col;
  v16h b;
#pragma unroll
  for (int i = 0; i < 8; ++i) {
    b[i]     = (_Float16)p[(size_t)i * ldn];
    b[i + 8] = (_Float16)p[(size_t)(i + 16) * ldn];
  }
  return b;
}

__device__ __forceinline__ v8f wmma_f16(v16h a, v16h b, v8f c) {
  return __builtin_amdgcn_wmma_f32_16x16x32_f16(false, a, false, b, (short)0, c,
                                                false, false);
}

// ---------------- CDNA5 async global -> LDS (per-lane 16B) ----------------
__device__ __forceinline__ void async_b128(unsigned lds_off, const void* gaddr) {
  asm volatile("global_load_async_to_lds_b128 %0, %1, off"
               :: "v"(lds_off), "v"((unsigned long long)gaddr) : "memory");
}
__device__ __forceinline__ void wait_async0() {
  asm volatile("s_wait_asynccnt 0" ::: "memory");
}

// deterministic hash -> uniform(0,1)
__device__ __forceinline__ float u01(unsigned a, unsigned b, unsigned c, unsigned d) {
  unsigned x = a * 0x9E3779B9u;
  x ^= (b + 0x7F4A7C15u) * 0x85EBCA6Bu;
  x ^= (c + 0x165667B1u) * 0xC2B2AE35u;
  x ^= (d + 0x27D4EB2Fu) * 0x9E3779B9u;
  x ^= x >> 16; x *= 0x7FEB352Du; x ^= x >> 15; x *= 0x846CA68Bu; x ^= x >> 16;
  return ((float)(x >> 8) + 0.5f) * (1.0f / 16777216.0f);
}

__global__ __launch_bounds__(256, 1)
void hier_ptr_decoder(
    const float* __restrict__ node_context, const float* __restrict__ cell_context,
    const float* __restrict__ original_data, const float* __restrict__ high_mask,
    const float* __restrict__ low_mask,     const float* __restrict__ init_w,
    const float* __restrict__ W_hc, const float* __restrict__ b_hc,
    const float* __restrict__ W_vw, const float* __restrict__ b_vw,
    const float* __restrict__ Wq,   const float* __restrict__ bq,
    const float* __restrict__ Wk,   const float* __restrict__ bk,
    const float* __restrict__ v_vec,
    const float* __restrict__ low_init_w,
    const float* __restrict__ low_W_hc, const float* __restrict__ low_b_hc,
    const float* __restrict__ low_W_vw, const float* __restrict__ low_b_vw,
    const float* __restrict__ low_Wq,   const float* __restrict__ low_bq,
    const float* __restrict__ low_Wk,   const float* __restrict__ low_bk,
    const float* __restrict__ low_v_vec,
    float* __restrict__ out)
{
  // ---- LDS (~260 KB of the 320 KB/WGP) ----
  __shared__ __attribute__((aligned(16))) _Float16 s_keys[ROWS * NNODE * EDIM];  // 160KB (also holds high keys)
  __shared__ __attribute__((aligned(16))) _Float16 s_cc  [ROWS * NCELL * EDIM];  // 48KB f16 cell_context cache
  __shared__ __attribute__((aligned(16))) float    s_stage[ROWS * EDIM];         // 8KB A-tile staging (f32)
  __shared__ __attribute__((aligned(16))) _Float16 s_acat[ROWS * 2 * EDIM];      // 8KB [init_h | h]
  __shared__ __attribute__((aligned(16))) _Float16 s_q   [ROWS * EDIM];          // 4KB current low query (f16)
  __shared__ __attribute__((aligned(16))) _Float16 s_qhi [ROWS * EDIM];          // 4KB high query (f16)
  __shared__ __attribute__((aligned(16))) _Float16 s_inithi[ROWS * EDIM];        // 4KB high init_h
  __shared__ __attribute__((aligned(16))) float    s_qproj[ROWS * HDIM];         // 8KB
  __shared__ __attribute__((aligned(16))) float    s_hbl [ROWS * EDIM];          // 8KB low h_bar
  __shared__ __attribute__((aligned(16))) float    s_hbh [ROWS * EDIM];          // 8KB high h_bar
  __shared__ float s_qc[2 * EDIM];                 // init_w@W_vw+b_vw (hi | lo)
  __shared__ float s_v[HDIM], s_lv[HDIM];
  __shared__ float s_logits[ROWS * NNODE];
  __shared__ int   s_idx[ROWS], s_cidx[ROWS];
  __shared__ unsigned long long s_mask[ROWS], s_hmask[ROWS];
  __shared__ float s_clp[ROWS], s_crew[ROWS], s_llp[ROWS], s_lrew[ROWS];
  __shared__ float s_first[ROWS][2], s_prev[ROWS][2], s_plast[ROWS][2];

  const int tid  = threadIdx.x;
  const int lane = tid & 31;
  const int nb   = (tid >> 5) << 4;           // wave's N-tile base column (8 waves * 16)
  const int b0   = blockIdx.x * ROWS;
  const int colW = nb + (lane & 15);          // this lane's output column
  const int mtop = (lane >> 4) << 3;          // D fragment: row = r + mtop

  if (tid == 0) {                              // warm L2 for the hot weights
    __builtin_prefetch(low_Wq, 0, 1);
    __builtin_prefetch(low_W_vw, 0, 1);
    __builtin_prefetch(low_Wk, 0, 1);
  }

  // ---------------- one-time init ----------------
  for (int i = tid; i < ROWS * NCELL * EDIM; i += 256)
    s_cc[i] = (_Float16)cell_context[(size_t)b0 * NCELL * EDIM + i];
  for (int i = tid; i < HDIM; i += 256) { s_v[i] = v_vec[i]; s_lv[i] = low_v_vec[i]; }
  if (tid < ROWS) {
    unsigned long long m = 0;
    for (int j = 0; j < NCELL; ++j)
      if (high_mask[(size_t)(b0 + tid) * NCELL + j] > 0.f) m |= 1ull << j;
    s_hmask[tid] = m;
    s_clp[tid] = 0.f; s_crew[tid] = 0.f;
    s_plast[tid][0] = 0.f; s_plast[tid][1] = 0.f;
  }
  if (tid < EDIM) {                            // q0 constants: init_w @ W_vw + b_vw
    float a = b_vw[tid];
    for (int k = 0; k < 2 * EDIM; ++k) a += init_w[k] * W_vw[(size_t)k * EDIM + tid];
    s_qc[tid] = a;
  } else if (tid < 2 * EDIM) {
    int h = tid - EDIM;
    float a = low_b_vw[h];
    for (int k = 0; k < 2 * EDIM; ++k) a += low_init_w[k] * low_W_vw[(size_t)k * EDIM + h];
    s_qc[EDIM + h] = a;
  }
  __syncthreads();

  // high h_bar = mean(cell_context,1) @ W_hc + b_hc
  for (int i = tid; i < ROWS * EDIM; i += 256) {
    int b = i >> 7, h = i & 127;
    float s = 0.f;
    for (int j = 0; j < NCELL; ++j) s += (float)s_cc[(b * NCELL + j) * EDIM + h];
    s_stage[i] = s * (1.0f / NCELL);
  }
  __syncthreads();
  {
    v8f acc = {};
#pragma unroll
    for (int kc = 0; kc < 4; ++kc)
      acc = wmma_f16(lda_f32(s_stage, EDIM, kc * 32), ldb_g(W_hc, EDIM, nb, kc * 32), acc);
    float bias = b_hc[colW];
#pragma unroll
    for (int r = 0; r < 8; ++r) s_hbh[(r + mtop) * EDIM + colW] = acc[r] + bias;
  }
  __syncthreads();
  for (int i = tid; i < ROWS * EDIM; i += 256)
    s_qhi[i] = (_Float16)(s_hbh[i] + s_qc[i & 127]);
  __syncthreads();

  // ================= cell (high) loop =================
  for (int ci = 0; ci < NCELL; ++ci) {
    // ---- high keys = cell_context @ Wk + bk   (12 M-tiles, A straight from s_cc)
    {
      v16h bf[4];
#pragma unroll
      for (int kc = 0; kc < 4; ++kc) bf[kc] = ldb_g(Wk, HDIM, nb, kc * 32);
      float bias = bk[colW];
      for (int mt = 0; mt < (ROWS * NCELL) / 16; ++mt) {
        v8f acc = {};
#pragma unroll
        for (int kc = 0; kc < 4; ++kc)
          acc = wmma_f16(lda_f16(s_cc + mt * 16 * EDIM, EDIM, kc * 32), bf[kc], acc);
#pragma unroll
        for (int r = 0; r < 8; ++r)
          s_keys[(mt * 16 + r + mtop) * HDIM + colW] = (_Float16)(acc[r] + bias);
      }
    }
    // ---- high qproj = query_hi @ Wq + bq
    {
      v8f acc = {};
#pragma unroll
      for (int kc = 0; kc < 4; ++kc)
        acc = wmma_f16(lda_f16(s_qhi, EDIM, kc * 32), ldb_g(Wq, HDIM, nb, kc * 32), acc);
      float bias = bq[colW];
#pragma unroll
      for (int r = 0; r < 8; ++r) s_qproj[(r + mtop) * HDIM + colW] = acc[r] + bias;
    }
    __syncthreads();
    // ---- logits over cells
    for (int p = tid; p < ROWS * NCELL; p += 256) {
      int b = p / NCELL, j = p % NCELL;
      const float*    qp = s_qproj + b * HDIM;
      const _Float16* kp = s_keys + p * HDIM;
      float u = 0.f;
#pragma unroll 4
      for (int h = 0; h < HDIM; ++h) u += s_v[h] * tanhf(qp[h] + (float)kp[h]);
      bool msk = (s_hmask[b] >> j) & 1ull;
      s_logits[b * NNODE + j] = msk ? NEGV : CLIPV * tanhf(u);
    }
    __syncthreads();
    // ---- sample high (Gumbel argmax) + log prob
    if (tid < ROWS) {
      int b = tid;
      const float* lg = s_logits + b * NNODE;
      float mx = -1e30f;
      for (int j = 0; j < NCELL; ++j) mx = fmaxf(mx, lg[j]);
      float se = 0.f;
      for (int j = 0; j < NCELL; ++j) se += expf(lg[j] - mx);
      float lse = mx + logf(se);
      float best = -1e30f; int bi = 0;
      for (int j = 0; j < NCELL; ++j) {
        float g = -logf(-logf(u01(0x48A1u, (unsigned)ci, (unsigned)(b0 + b), (unsigned)j)));
        if (lg[j] + g > best) { best = lg[j] + g; bi = j; }
      }
      s_cidx[b] = bi;
      s_clp[b] += lg[bi] - lse;
      s_hmask[b] |= 1ull << bi;
      out[OFF_CACT + (size_t)(b0 + b) * NCELL + ci] = (float)bi;
    }
    __syncthreads();
    // ---- high query update: query = h_bar + [init_h|h] @ W_vw + b_vw
    for (int i = tid; i < ROWS * EDIM; i += 256) {
      int b = i >> 7, h = i & 127;
      _Float16 hv = s_cc[(b * NCELL + s_cidx[b]) * EDIM + h];
      if (ci == 0) s_inithi[i] = hv;
      s_acat[b * (2 * EDIM) + h]        = s_inithi[i];
      s_acat[b * (2 * EDIM) + EDIM + h] = hv;
    }
    __syncthreads();
    {
      v8f acc = {};
#pragma unroll
      for (int kc = 0; kc < 8; ++kc)
        acc = wmma_f16(lda_f16(s_acat, 2 * EDIM, kc * 32), ldb_g(W_vw, EDIM, nb, kc * 32), acc);
      float bias = b_vw[colW];
#pragma unroll
      for (int r = 0; r < 8; ++r) {
        int m = r + mtop;
        s_qhi[m * EDIM + colW] = (_Float16)(acc[r] + bias + s_hbh[m * EDIM + colW]);
      }
    }
    __syncthreads();

    // ================= low decode for the chosen cell =================
    // low h_bar = mean(cell_emb,1) @ low_W_hc + low_b_hc
    for (int i = tid; i < ROWS * EDIM; i += 256) {
      int b = i >> 7, h = i & 127;
      const float* src = node_context +
          ((((size_t)(b0 + b)) * NCELL + s_cidx[b]) * NNODE) * EDIM + h;
      float s = 0.f;
      for (int n = 0; n < NNODE; ++n) s += src[(size_t)n * EDIM];
      s_stage[i] = s * (1.0f / NNODE);
    }
    __syncthreads();
    {
      v8f acc = {};
#pragma unroll
      for (int kc = 0; kc < 4; ++kc)
        acc = wmma_f16(lda_f32(s_stage, EDIM, kc * 32), ldb_g(low_W_hc, EDIM, nb, kc * 32), acc);
      float bias = low_b_hc[colW];
#pragma unroll
      for (int r = 0; r < 8; ++r) s_hbl[(r + mtop) * EDIM + colW] = acc[r] + bias;
    }
    __syncthreads();

    // low keys = cell_emb @ low_Wk + low_bk : async-stage gathered rows, 40 M-tiles
    {
      v16h bf[4];
#pragma unroll
      for (int kc = 0; kc < 4; ++kc) bf[kc] = ldb_g(low_Wk, HDIM, nb, kc * 32);
      float bias = low_bk[colW];
      for (int mt = 0; mt < (ROWS * NNODE) / 16; ++mt) {
#pragma unroll
        for (int c = 0; c < 2; ++c) {          // 512 x 16B chunks, 2 per thread
          int chunk = tid + c * 256;
          int r = chunk >> 5, sub = chunk & 31;
          int g = mt * 16 + r;
          int b = g / NNODE, n = g % NNODE;
          const float* gp = node_context +
              ((((size_t)(b0 + b)) * NCELL + s_cidx[b]) * NNODE + n) * EDIM + sub * 4;
          async_b128((unsigned)(unsigned long long)(&s_stage[r * EDIM + sub * 4]), gp);
        }
        wait_async0();
        __syncthreads();
        v8f acc = {};
#pragma unroll
        for (int kc = 0; kc < 4; ++kc)
          acc = wmma_f16(lda_f32(s_stage, EDIM, kc * 32), bf[kc], acc);
#pragma unroll
        for (int r = 0; r < 8; ++r)
          s_keys[(mt * 16 + r + mtop) * HDIM + colW] = (_Float16)(acc[r] + bias);
        __syncthreads();
      }
    }

    // low decode init
    if (tid < ROWS) {
      int b = tid;
      unsigned long long m = 0;
      for (int n = 0; n < NNODE; ++n)
        if (low_mask[(size_t)(b0 + b) * NNODE + n] > 0.f) m |= 1ull << n;
      s_mask[b] = m; s_llp[b] = 0.f; s_lrew[b] = 0.f;
    }
    for (int i = tid; i < ROWS * EDIM; i += 256)
      s_q[i] = (_Float16)(s_hbl[i] + s_qc[EDIM + (i & 127)]);
    __syncthreads();

    // hoist low-level B fragments into VGPRs for the whole 40-step loop
    v16h bqf[4], bwf[8];
#pragma unroll
    for (int kc = 0; kc < 4; ++kc) bqf[kc] = ldb_g(low_Wq, HDIM, nb, kc * 32);
#pragma unroll
    for (int kc = 0; kc < 8; ++kc) bwf[kc] = ldb_g(low_W_vw, EDIM, nb, kc * 32);
    const float lbq = low_bq[colW];
    const float lbw = low_b_vw[colW];

    for (int t = 0; t < NNODE; ++t) {
      // qproj = q @ low_Wq + low_bq
      {
        v8f acc = {};
#pragma unroll
        for (int kc = 0; kc < 4; ++kc)
          acc = wmma_f16(lda_f16(s_q, EDIM, kc * 32), bqf[kc], acc);
#pragma unroll
        for (int r = 0; r < 8; ++r) s_qproj[(r + mtop) * HDIM + colW] = acc[r] + lbq;
      }
      __syncthreads();
      // logits over 640 (b,n) pairs
      for (int p = tid; p < ROWS * NNODE; p += 256) {
        int b = p / NNODE, n = p % NNODE;
        const float*    qp = s_qproj + b * HDIM;
        const _Float16* kp = s_keys + p * HDIM;
        float u = 0.f;
#pragma unroll 4
        for (int h = 0; h < HDIM; ++h) u += s_lv[h] * tanhf(qp[h] + (float)kp[h]);
        bool msk = (s_mask[b] >> n) & 1ull;
        s_logits[p] = msk ? NEGV : CLIPV * tanhf(u);
      }
      __syncthreads();
      // sample low + rewards
      if (tid < ROWS) {
        int b = tid;
        const float* lg = s_logits + b * NNODE;
        float mx = -1e30f;
        for (int n = 0; n < NNODE; ++n) mx = fmaxf(mx, lg[n]);
        float se = 0.f;
        for (int n = 0; n < NNODE; ++n) se += expf(lg[n] - mx);
        float lse = mx + logf(se);
        float best = -1e30f; int bi = 0;
        for (int n = 0; n < NNODE; ++n) {
          float g = -logf(-logf(u01(0xB10Cu + (unsigned)t, (unsigned)ci,
                                    (unsigned)(b0 + b), (unsigned)n)));
          if (lg[n] + g > best) { best = lg[n] + g; bi = n; }
        }
        s_idx[b] = bi;
        s_llp[b] += lg[bi] - lse;
        s_mask[b] |= 1ull << bi;
        const float* cp = original_data +
            ((((size_t)(b0 + b)) * NCELL + s_cidx[b]) * NNODE + bi) * 2;
        float cx = cp[0], cy = cp[1];
        if (t == 0) {
          s_first[b][0] = cx; s_first[b][1] = cy;
          if (ci > 0) {
            float dx = cx - s_plast[b][0], dy = cy - s_plast[b][1];
            s_crew[b] += sqrtf(dx * dx + dy * dy);
          }
        } else {
          float dx = cx - s_prev[b][0], dy = cy - s_prev[b][1];
          s_lrew[b] += sqrtf(dx * dx + dy * dy);
        }
        s_prev[b][0] = cx; s_prev[b][1] = cy;
        if (t == NNODE - 1) { s_plast[b][0] = cx; s_plast[b][1] = cy; }
        out[OFF_NACT + ((size_t)(b0 + b) * NCELL + ci) * NNODE + t] = (float)bi;
      }
      __syncthreads();
      // gather h = cell_emb[b, idx] into Acat
      for (int i = tid; i < ROWS * EDIM; i += 256) {
        int b = i >> 7, h = i & 127;
        const float* hp = node_context +
            ((((size_t)(b0 + b)) * NCELL + s_cidx[b]) * NNODE + s_idx[b]) * EDIM;
        _Float16 hv = (_Float16)hp[h];
        if (t == 0) s_acat[b * (2 * EDIM) + h] = hv;
        s_acat[b * (2 * EDIM) + EDIM + h] = hv;
      }
      __syncthreads();
      // query = h_bar + Acat @ low_W_vw + low_b_vw
      {
        v8f acc = {};
#pragma unroll
        for (int kc = 0; kc < 8; ++kc)
          acc = wmma_f16(lda_f16(s_acat, 2 * EDIM, kc * 32), bwf[kc], acc);
#pragma unroll
        for (int r = 0; r < 8; ++r) {
          int m = r + mtop;
          s_q[m * EDIM + colW] = (_Float16)(acc[r] + lbw + s_hbl[m * EDIM + colW]);
        }
      }
      __syncthreads();
    } // node steps

    if (tid < ROWS) {
      out[OFF_NLP + (size_t)(b0 + tid) * NCELL + ci] = s_llp[tid];
      out[OFF_NRW + (size_t)(b0 + tid) * NCELL + ci] = s_lrew[tid];
    }
    __syncthreads();
  } // cells

  if (tid < ROWS) {
    out[OFF_CLP + b0 + tid] = s_clp[tid];
    out[OFF_CRW + b0 + tid] = s_crew[tid];
  }
}

extern "C" void kernel_launch(void* const* d_in, const int* in_sizes, int n_in,
                              void* d_out, int out_size, void* d_ws, size_t ws_size,
                              hipStream_t stream) {
  (void)in_sizes; (void)n_in; (void)out_size; (void)d_ws; (void)ws_size;
  hier_ptr_decoder<<<dim3(B_TOT / ROWS), dim3(256), 0, stream>>>(
      (const float*)d_in[0],  (const float*)d_in[1],  (const float*)d_in[2],
      (const float*)d_in[3],  (const float*)d_in[4],  (const float*)d_in[5],
      (const float*)d_in[6],  (const float*)d_in[7],  (const float*)d_in[8],
      (const float*)d_in[9],  (const float*)d_in[10], (const float*)d_in[11],
      (const float*)d_in[12], (const float*)d_in[13], (const float*)d_in[14],
      (const float*)d_in[15], (const float*)d_in[16], (const float*)d_in[17],
      (const float*)d_in[18], (const float*)d_in[19], (const float*)d_in[20],
      (const float*)d_in[21], (const float*)d_in[22], (const float*)d_in[23],
      (const float*)d_in[24], (float*)d_out);
}
